// PoseCDE_66726611911359
// MI455X (gfx1250) — matrix operand order repaired and checked by reading, not compiled, not run on hardware
//
#include <hip/hip_runtime.h>
#include <cmath>
#include <cstdint>

// ---------------- constants ----------------
#define BATCH 1024
#define LSEQ  11
#define HDIM  128
#define VF_   512
#define IF_   256
#define FDIM  768
#define CDIM  129
#define CSPLIT 16
#define MROWS  64          // batch rows per cde_field block

typedef __attribute__((ext_vector_type(16))) _Float16 v16h;
typedef __attribute__((ext_vector_type(8)))  _Float16 v8h;
typedef __attribute__((ext_vector_type(8)))  float    v8f;

// ---------------- WMMA helpers ----------------
// A matrix 16x32 f16 (M x K), row-major source with leading dim 128 halves.
// lanes 0-15: M=lane, K [kk*32 + 0..15]
// lanes 16-31: M=lane-16, K [kk*32+16..31]
__device__ __forceinline__ v16h a_frag(const _Float16* A, int lane, int kk) {
    int m  = lane & 15;
    int hf = lane >> 4;
    const v8h lo = *(const v8h*)(A + m * 128 + kk * 32 + hf * 16);
    const v8h hi = *(const v8h*)(A + m * 128 + kk * 32 + hf * 16 + 8);
    v16h r;
#pragma unroll
    for (int i = 0; i < 8; ++i) { r[i] = lo[i]; r[8 + i] = hi[i]; }
    return r;
}

// B matrix 32x16 f16 (K x N) sourced from transposed weights Wt[n][k] (k contiguous).
// lane: N = n0 + (lane&15); K chunk = kk*32 + (lane>>4)*16 .. +15 (contiguous 32B).
__device__ __forceinline__ v16h b_frag(const _Float16* Wt, int n0, int lane, int kk) {
    int col = lane & 15;
    int hf  = lane >> 4;
    return *(const v16h*)(Wt + (size_t)(n0 + col) * 128 + kk * 32 + hf * 16);
}

__device__ __forceinline__ v8f wmma16(v16h a, v16h b, v8f c) {
    return __builtin_amdgcn_wmma_f32_16x16x32_f16(false, a, false, b, (short)0, c, false, false);
}

// ---------------- prep: weight transpose/convert ----------------
__global__ void cde_prep_t(const float* __restrict__ w, _Float16* __restrict__ wt) {
    int i = blockIdx.x * blockDim.x + threadIdx.x;          // 16384
    int n = i >> 7, k = i & 127;
    wt[n * 128 + k] = (_Float16)w[k * 128 + n];
}

// w_out [128][16512] -> wT[c][h][k]  (129*128*128 f16)
__global__ void cde_prep_wout(const float* __restrict__ w, _Float16* __restrict__ wt) {
    int i = blockIdx.x * blockDim.x + threadIdx.x;
    if (i >= CDIM * 128 * 128) return;
    int k = i & 127;
    int t = i >> 7;
    int h = t & 127;
    int c = t >> 7;
    wt[i] = (_Float16)w[(size_t)k * 16512 + h * 129 + c];
}

// ---------------- reduction net: fused[:,0:3,:] ----------------
__global__ __launch_bounds__(128) void cde_reduce(const float* __restrict__ fv,
                                                  const float* __restrict__ fi,
                                                  const float* __restrict__ wred,
                                                  const float* __restrict__ bred,
                                                  float* __restrict__ fused3) {
    __shared__ float xin[FDIM];
    int b = blockIdx.x / 3, l = blockIdx.x % 3;
    int tid = threadIdx.x;
    const float* pv = fv + ((size_t)b * LSEQ + l) * VF_;
    const float* pi = fi + ((size_t)b * LSEQ + l) * IF_;
    for (int i = tid; i < VF_; i += 128) xin[i] = pv[i];
    for (int i = tid; i < IF_; i += 128) xin[VF_ + i] = pi[i];
    __syncthreads();
    float s = bred[tid];
    for (int k = 0; k < FDIM; ++k) s += xin[k] * wred[k * 128 + tid];
    fused3[((size_t)b * 3 + l) * 128 + tid] = s;
}

// ---------------- init: z0 and dX0/dX1 ----------------
__global__ __launch_bounds__(128) void cde_init(const float* __restrict__ ts,
                                                const float* __restrict__ fused3,
                                                const float* __restrict__ winit,
                                                const float* __restrict__ binit,
                                                float* __restrict__ z, float* __restrict__ zs,
                                                float* __restrict__ hT, float* __restrict__ dX) {
    __shared__ float x0[CDIM];
    int b = blockIdx.x, tid = threadIdx.x;
    const float* tsb = ts + (size_t)b * (LSEQ + 1);
    const float* fb  = fused3 + (size_t)b * 384;
    if (tid == 0) x0[0] = tsb[1] - tsb[0];
    x0[1 + tid] = fb[tid];
    __syncthreads();
    float s = binit[tid];
    for (int c = 0; c < CDIM; ++c) s += x0[c] * winit[c * 128 + tid];
    z[(size_t)b * 128 + tid]  = s;
    zs[(size_t)b * 128 + tid] = s;
    hT[(size_t)b * 1280 + tid] = s;                         // hT[:,0,:] = z0
    for (int c = tid; c < CDIM; c += 128) {
        float d0, d1;
        if (c == 0) { d0 = tsb[2] - tsb[1]; d1 = tsb[3] - tsb[2]; }
        else {
            float f0 = fb[c - 1], f1 = fb[128 + c - 1], f2 = fb[256 + c - 1];
            d0 = f1 - f0; d1 = f2 - f1;
        }
        dX[(size_t)b * 258 + c]        = d0;
        dX[(size_t)b * 258 + 129 + c]  = d1;
    }
}

// ---------------- stage front: h2 = relu(relu(z@W1+b1)@W2+b2)  (f16 out) ----------------
__global__ __launch_bounds__(256) void cde_mlp(const float* __restrict__ zs,
                                               const _Float16* __restrict__ tw1,
                                               const float* __restrict__ b1,
                                               const _Float16* __restrict__ tw2,
                                               const float* __restrict__ b2,
                                               _Float16* __restrict__ h2buf) {
    __shared__ __align__(32) _Float16 Ash[16 * 128];
    __shared__ __align__(32) _Float16 Hsh[16 * 128];
    int b0 = blockIdx.x * 16;
    int tid = threadIdx.x;
    for (int i = tid; i < 2048; i += 256) Ash[i] = (_Float16)zs[(size_t)b0 * 128 + i];
    __syncthreads();
    int lane = tid & 31, wave = tid >> 5;
    int n0 = wave * 16;
    int hf = lane >> 4;
    int ncol = n0 + (lane & 15);

    v8f acc = {};
#pragma unroll
    for (int kk = 0; kk < 4; ++kk)
        acc = wmma16(a_frag(Ash, lane, kk), b_frag(tw1, n0, lane, kk), acc);
    float bias = b1[ncol];
#pragma unroll
    for (int v = 0; v < 8; ++v) {
        float x = acc[v] + bias;
        Hsh[(hf * 8 + v) * 128 + ncol] = (_Float16)(x > 0.f ? x : 0.f);
    }
    __syncthreads();

    v8f acc2 = {};
#pragma unroll
    for (int kk = 0; kk < 4; ++kk)
        acc2 = wmma16(a_frag(Hsh, lane, kk), b_frag(tw2, n0, lane, kk), acc2);
    float bias2 = b2[ncol];
#pragma unroll
    for (int v = 0; v < 8; ++v) {
        float x = acc2[v] + bias2;
        Ash[(hf * 8 + v) * 128 + ncol] = (_Float16)(x > 0.f ? x : 0.f);
    }
    __syncthreads();
    for (int i = tid; i < 256; i += 256)
        ((v8h*)h2buf)[(size_t)b0 * 16 + i] = ((const v8h*)Ash)[i];
}

// ---------------- stage main: vf partial = sum_c tanh(h2@Wc+b)*dX[:,c] ----------------
// Block: 512 threads = 16 waves; 64 batch rows x 128 h cols.
// waves 0-7: rows [0,32) cols 16w ; waves 8-15: rows [32,64) cols 16(w-8).
// Each wave owns two 16-row tiles -> every B fragment feeds 2 WMMAs.
// B fragments read directly from global (WGP$/L2-hot slab), next slab prefetched.
__global__ __launch_bounds__(512) void cde_field(const _Float16* __restrict__ h2buf,
                                                 const _Float16* __restrict__ twout,
                                                 const float* __restrict__ bout,
                                                 const float* __restrict__ dX, int idx,
                                                 float* __restrict__ kpart) {
    __shared__ __align__(32) _Float16 Ash[MROWS * 128];     // 16KB
    __shared__ float dXs[MROWS * CDIM];                     // 33KB
    int b0 = blockIdx.x * MROWS;
    int cs = blockIdx.y;
    int tid = threadIdx.x;
    for (int i = tid; i < MROWS * 128 / 8; i += 512)
        ((v8h*)Ash)[i] = ((const v8h*)h2buf)[(size_t)b0 * 16 + i];
    for (int i = tid; i < MROWS * CDIM; i += 512) {
        int r = i / CDIM, c = i % CDIM;
        dXs[i] = dX[(size_t)(b0 + r) * 258 + idx * 129 + c];
    }
    __syncthreads();
    int lane = tid & 31, wave = tid >> 5;
    int rbase = (wave >> 3) * 32;                           // 0 or 32
    int n0 = (wave & 7) * 16;
    int hf = lane >> 4;
    int ncol = n0 + (lane & 15);

    v16h a0[4], a1[4];
#pragma unroll
    for (int kk = 0; kk < 4; ++kk) {
        a0[kk] = a_frag(Ash + rbase * 128, lane, kk);
        a1[kk] = a_frag(Ash + (rbase + 16) * 128, lane, kk);
    }

    float f0[8], f1[8];
#pragma unroll
    for (int v = 0; v < 8; ++v) { f0[v] = 0.f; f1[v] = 0.f; }

    for (int c = cs; c < CDIM; c += CSPLIT) {
        const _Float16* slab = twout + (size_t)c * 128 * 128;
        int cn = c + CSPLIT;
        if (cn < CDIM)                                      // pull next 32KB slab toward WGP
            __builtin_prefetch((const char*)(twout + (size_t)cn * 128 * 128) + tid * 64, 0, 3);
        v8f acc0 = {}, acc1 = {};
#pragma unroll
        for (int kk = 0; kk < 4; ++kk) {
            v16h bb = b_frag(slab, n0, lane, kk);
            acc0 = wmma16(a0[kk], bb, acc0);
            acc1 = wmma16(a1[kk], bb, acc1);
        }
        float bo = bout[ncol * 129 + c];
#pragma unroll
        for (int v = 0; v < 8; ++v) {
            f0[v] += tanhf(acc0[v] + bo) * dXs[(rbase + hf * 8 + v) * CDIM + c];
            f1[v] += tanhf(acc1[v] + bo) * dXs[(rbase + 16 + hf * 8 + v) * CDIM + c];
        }
    }
#pragma unroll
    for (int v = 0; v < 8; ++v) {
        kpart[((size_t)cs * BATCH + b0 + rbase + hf * 8 + v) * 128 + ncol]      = f0[v];
        kpart[((size_t)cs * BATCH + b0 + rbase + 16 + hf * 8 + v) * 128 + ncol] = f1[v];
    }
}

// ---------------- RK4 bookkeeping + partial reduce ----------------
__global__ __launch_bounds__(256) void cde_rk(float* __restrict__ z, float* __restrict__ zs,
                                              float* __restrict__ kacc,
                                              const float* __restrict__ kpart,
                                              float* __restrict__ hT,
                                              int mode, float h, int slot) {
    int i = blockIdx.x * blockDim.x + threadIdx.x;          // BATCH*HDIM
    const int P = BATCH * HDIM;
    float k = 0.f;
#pragma unroll
    for (int p = 0; p < CSPLIT; ++p) k += kpart[(size_t)p * P + i];
    if (mode == 0)      { kacc[i] = k;          zs[i] = z[i] + 0.5f * h * k; }
    else if (mode == 1) { kacc[i] += 2.f * k;   zs[i] = z[i] + 0.5f * h * k; }
    else if (mode == 2) { kacc[i] += 2.f * k;   zs[i] = z[i] + h * k; }
    else {
        float zn = z[i] + (h / 6.f) * (kacc[i] + k);
        z[i] = zn; zs[i] = zn;
        if (slot >= 0) {
            int b = i >> 7, hh = i & 127;
            hT[((size_t)b * 10 + slot) * 128 + hh] = zn;
        }
    }
}

// ---------------- readout: leaky(hT@Wr1+b)@Wr2+b ----------------
__global__ __launch_bounds__(256) void cde_readout(const float* __restrict__ hT,
                                                   const _Float16* __restrict__ twr1,
                                                   const float* __restrict__ br1,
                                                   const float* __restrict__ wr2,
                                                   const float* __restrict__ br2,
                                                   float* __restrict__ out) {
    __shared__ __align__(32) _Float16 Ash[16 * 128];
    __shared__ float Asf[16 * 128];
    int r0 = blockIdx.x * 16;                               // rows of [10240,128]
    int tid = threadIdx.x;
    for (int i = tid; i < 2048; i += 256) Ash[i] = (_Float16)hT[(size_t)r0 * 128 + i];
    __syncthreads();
    int lane = tid & 31, wave = tid >> 5;
    int n0 = wave * 16;
    int hf = lane >> 4;
    int ncol = n0 + (lane & 15);
    v8f acc = {};
#pragma unroll
    for (int kk = 0; kk < 4; ++kk)
        acc = wmma16(a_frag(Ash, lane, kk), b_frag(twr1, n0, lane, kk), acc);
    float bias = br1[ncol];
#pragma unroll
    for (int v = 0; v < 8; ++v) {
        float x = acc[v] + bias;
        Asf[(hf * 8 + v) * 128 + ncol] = (x > 0.f ? x : 0.1f * x);
    }
    __syncthreads();
    if (tid < 96) {
        int r = tid / 6, j = tid % 6;
        float s = br2[j];
        for (int k = 0; k < 128; ++k) s += Asf[r * 128 + k] * wr2[k * 6 + j];
        out[(size_t)(r0 + r) * 6 + j] = s;
    }
}

__global__ void cde_hlast(const float* __restrict__ hT, float* __restrict__ out) {
    int i = blockIdx.x * blockDim.x + threadIdx.x;          // BATCH*HDIM
    int b = i >> 7, hh = i & 127;
    out[61440 + i] = hT[((size_t)b * 10 + 9) * 128 + hh];
}

// ---------------- host ----------------
extern "C" void kernel_launch(void* const* d_in, const int* in_sizes, int n_in,
                              void* d_out, int out_size, void* d_ws, size_t ws_size,
                              hipStream_t stream) {
    const float* fv    = (const float*)d_in[0];
    const float* fi    = (const float*)d_in[1];
    const float* ts    = (const float*)d_in[2];
    const float* w_red = (const float*)d_in[3];
    const float* b_red = (const float*)d_in[4];
    const float* w_init= (const float*)d_in[5];
    const float* b_init= (const float*)d_in[6];
    const float* w_h1  = (const float*)d_in[7];
    const float* b_h1  = (const float*)d_in[8];
    const float* w_h2  = (const float*)d_in[9];
    const float* b_h2  = (const float*)d_in[10];
    const float* w_out = (const float*)d_in[11];
    const float* b_out = (const float*)d_in[12];
    const float* w_r1  = (const float*)d_in[13];
    const float* b_r1  = (const float*)d_in[14];
    const float* w_r2  = (const float*)d_in[15];
    const float* b_r2  = (const float*)d_in[16];
    float* out = (float*)d_out;

    char* ws = (char*)d_ws;
    _Float16* tw1    = (_Float16*)(ws + 0);
    _Float16* tw2    = (_Float16*)(ws + 32768);
    _Float16* twr1   = (_Float16*)(ws + 65536);
    _Float16* twout  = (_Float16*)(ws + 98304);            // 4,227,072 B
    float*    fused3 = (float*)(ws + 4325376);             // 1,572,864 B
    float*    dX     = (float*)(ws + 5898240);             // 1,056,768 B
    float*    z      = (float*)(ws + 6955008);
    float*    zsbuf  = (float*)(ws + 7479296);
    float*    kacc   = (float*)(ws + 8003584);
    _Float16* h2buf  = (_Float16*)(ws + 8527872);          // 262,144 B
    float*    kpart  = (float*)(ws + 8790016);             // 8,388,608 B (16 planes)
    float*    hT     = (float*)(ws + 17178624);            // 5,242,880 B -> end 22,421,504

    // ---- weight prep ----
    cde_prep_t<<<64, 256, 0, stream>>>(w_h1, tw1);
    cde_prep_t<<<64, 256, 0, stream>>>(w_h2, tw2);
    cde_prep_t<<<64, 256, 0, stream>>>(w_r1, twr1);
    cde_prep_wout<<<(CDIM * 128 * 128 + 255) / 256, 256, 0, stream>>>(w_out, twout);

    // ---- reduction + init ----
    cde_reduce<<<BATCH * 3, 128, 0, stream>>>(fv, fi, w_red, b_red, fused3);
    cde_init<<<BATCH, 128, 0, stream>>>(ts, fused3, w_init, b_init, z, zsbuf, hT, dX);

    // ---- ODE loop: 9 intervals x 4 substeps x 4 RK stages ----
    const float step  = (1.0f - 0.1f) / 9.0f;               // linspace delta
    const float et1   = 0.1f + 1.0f * step;
    const float hstep = (et1 - 0.1f) / 4.0f;                // SUBSTEPS = 4
    for (int i = 0; i < 9; ++i) {
        float t0i = 0.1f + (float)i * step;
        for (int j = 0; j < 4; ++j) {
            float t = t0i + (float)j * hstep;
            float tstage[4] = { t, t + hstep * 0.5f, t + hstep * 0.5f, t + hstep };
            for (int s = 0; s < 4; ++s) {
                int idx = (int)floorf(tstage[s]);
                if (idx < 0) idx = 0;
                if (idx > 9) idx = 9;
                if (idx > 1) idx = 1;                       // only 2 dX planes are reachable
                cde_mlp<<<BATCH / 16, 256, 0, stream>>>(zsbuf, tw1, b_h1, tw2, b_h2, h2buf);
                cde_field<<<dim3(BATCH / MROWS, CSPLIT, 1), 512, 0, stream>>>(
                    h2buf, twout, b_out, dX, idx, kpart);
                int slot = (s == 3 && j == 3) ? (i + 1) : -1;
                cde_rk<<<(BATCH * HDIM) / 256, 256, 0, stream>>>(
                    z, zsbuf, kacc, kpart, hT, s, hstep, slot);
            }
        }
    }

    // ---- readout ----
    cde_readout<<<(BATCH * 10) / 16, 256, 0, stream>>>(hT, twr1, b_r1, w_r2, b_r2, out);
    cde_hlast<<<(BATCH * HDIM) / 256, 256, 0, stream>>>(hT, out);

    (void)in_sizes; (void)n_in; (void)out_size; (void)ws_size;
}
